// TemporalMultiHeadAttention_18107582120299
// MI455X (gfx1250) — matrix operand order repaired and checked by reading, not compiled, not run on hardware
//
#include <hip/hip_runtime.h>
#include <hip/hip_bf16.h>

typedef __attribute__((ext_vector_type(16))) __bf16 v16bf;
typedef __attribute__((ext_vector_type(8)))  float  v8f;
typedef __attribute__((ext_vector_type(4)))  unsigned int v4u;
typedef __attribute__((ext_vector_type(8)))  int v8i;
typedef __attribute__((ext_vector_type(4)))  int v4i;

#define BB 2
#define SS 2048
#define DD 1024
#define HH 16
#define HDD 64

// TDM: builtin exists on both toolchains, but with different arity:
//   clang-22 (ROCm 7.2):        5 args (v4u, v8i, v4i, v4i, i32)
//   clang-23 (amdgpu-toolchain): 6 args (v4u, v8i, v4i, v4i, v8i, i32)
#if defined(__AMDGCN__) && \
    __has_builtin(__builtin_amdgcn_tensor_load_to_lds) && \
    __has_builtin(__builtin_amdgcn_s_wait_tensorcnt)
#define USE_TDM 1
#if (__clang_major__ >= 23)
#define TDM_ARGS6 1
#else
#define TDM_ARGS6 0
#endif
#else
#define USE_TDM 0
#endif

struct bf16x16_raw { uint4 lo, hi; };

// A-fragment (16x32 bf16): lane l holds row m=l&15; elements K = 2*kb..2*kb+15,
// kb = (l>>4)*8.  Two contiguous 16B chunks.
static __device__ inline v16bf load_a_frag(const __bf16* row, int kb) {
    bf16x16_raw r;
    r.lo = *(const uint4*)(row + 2 * kb);
    r.hi = *(const uint4*)(row + 2 * kb + 8);
    return __builtin_bit_cast(v16bf, r);
}

// B-fragment (32x16 bf16) from a K-contiguous buffer: lane l holds col n=l&15,
// elements K = kb16..kb16+15 contiguous -> one 32B chunk.
static __device__ inline v16bf load_b_frag_contig(const __bf16* p) {
    return *(const v16bf*)p;
}

static __device__ inline v8f wmma_bf16(v16bf a, v16bf b, v8f c) {
    return __builtin_amdgcn_wmma_f32_16x16x32_bf16(false, a, false, b, (short)0, c, false, false);
}

static __device__ inline v8f v8f_zero() {
    v8f z = {0.f,0.f,0.f,0.f,0.f,0.f,0.f,0.f};
    return z;
}

#if USE_TDM
// 2-D tensor_load_to_lds: tile_d0 x tile_d1 bf16 elements, dim0 contiguous in memory,
// rows stride0 elements apart; LDS rows padded 128B data + 32B pad (stride 80 elems).
// D# packing per CDNA5 ISA.
static __device__ inline void tdm_load_tile_2d(unsigned lds_off, const __bf16* g,
                                               unsigned tile_d0, unsigned tile_d1,
                                               unsigned long long stride0_elems) {
    const unsigned long long ga = (unsigned long long)(uintptr_t)g;
    v4u g0;
    g0[0] = 1u;                                   // count=1, user descriptor
    g0[1] = lds_off;                              // lds_addr (bytes)
    g0[2] = (unsigned)ga;                         // global_addr[31:0]
    g0[3] = (unsigned)((ga >> 32) & 0x01ffffffu)  // global_addr[56:32]
          | (2u << 30);                           // type=2 ("image")
    v8i g1;
    g1[0] = (int)((1u << 16)      // data_size = 2 bytes
                | (1u << 20)      // pad_enable
                | (4u << 22)      // pad_interval: 32 dwords = 128B
                | (7u << 25));    // pad_amount:   8 dwords = 32B
    g1[1] = (int)((tile_d0 & 0xffffu) << 16);                   // tensor_dim0[15:0]
    g1[2] = (int)(((tile_d0 >> 16) & 0xffffu)
                | ((tile_d1 & 0xffffu) << 16));                 // tensor_dim0 hi | tensor_dim1 lo
    g1[3] = (int)(((tile_d1 >> 16) & 0xffffu)
                | ((tile_d0 & 0xffffu) << 16));                 // tensor_dim1 hi | tile_dim0
    g1[4] = (int)(tile_d1 & 0xffffu);                           // tile_dim1 (tile_dim2=0 -> 2D)
    g1[5] = (int)(unsigned)(stride0_elems & 0xffffffffull);     // tensor_dim0_stride[31:0]
    g1[6] = (int)(unsigned)((stride0_elems >> 32) & 0xffffull); // stride hi | stride1=0
    g1[7] = 0;
    v4i gz = {0, 0, 0, 0};
#if TDM_ARGS6
    v8i gz8 = {0, 0, 0, 0, 0, 0, 0, 0};
    __builtin_amdgcn_tensor_load_to_lds(g0, g1, gz, gz, gz8, 0);
#else
    __builtin_amdgcn_tensor_load_to_lds(g0, g1, gz, gz, 0);
#endif
}
#endif

__global__ void f32_to_bf16_kernel(const float* __restrict__ in, __bf16* __restrict__ out, int n) {
    int i = blockIdx.x * blockDim.x + threadIdx.x;
    if (i < n) out[i] = (__bf16)in[i];
}

// W[K][N] f32 -> Wt[N][K] bf16 (one-time; reads coalesced).
__global__ void transpose_f32_to_bf16_kernel(const float* __restrict__ in, __bf16* __restrict__ out,
                                             int K, int N) {
    int i = blockIdx.x * blockDim.x + threadIdx.x;
    if (i < K * N) {
        int k = i / N, n = i - k * N;
        out[(size_t)n * K + k] = (__bf16)in[i];
    }
}

// C = A[M,K] * B + bias, B given TRANSPOSED: Bt[N][K] (bf16).
// MODE 0: row-major C.  MODE 1: V-layout out[((row>>11)*1024+col)*2048 + (row&2047)].
template <typename OutT, int MODE>
__global__ void __launch_bounds__(128)
gemm_bt_bias_kernel(const __bf16* __restrict__ A, const __bf16* __restrict__ Bt,
                    const float* __restrict__ bias, OutT* __restrict__ C,
                    int M, int N, int K) {
    __shared__ __bf16 As[128 * 40];
    __shared__ __bf16 Bs[128 * 40];

    const int t    = threadIdx.x;
    const int lane = t & 31;
    const int wave = t >> 5;
    const int wm0  = (wave >> 1) * 64;
    const int wn0  = (wave & 1) * 64;
    const int rowg0 = blockIdx.x * 128;
    const int colg0 = blockIdx.y * 128;
    const int ln   = lane & 15;
    const int kbA  = (lane >> 4) * 8;
    const int kbB  = (lane >> 4) * 16;

    v8f acc[4][4];
#pragma unroll
    for (int i = 0; i < 4; i++)
#pragma unroll
        for (int j = 0; j < 4; j++) acc[i][j] = v8f_zero();

    for (int k0 = 0; k0 < K; k0 += 32) {
        {
            const __bf16* g = A + (size_t)(rowg0 + t) * K + k0;
#pragma unroll
            for (int i = 0; i < 4; i++)
                *(uint4*)&As[t * 40 + i * 8] = *(const uint4*)(g + i * 8);
        }
        {
            const __bf16* g = Bt + (size_t)(colg0 + t) * K + k0;
#pragma unroll
            for (int i = 0; i < 4; i++)
                *(uint4*)&Bs[t * 40 + i * 8] = *(const uint4*)(g + i * 8);
        }
        __syncthreads();

        v16bf af[4], bfrag[4];
#pragma unroll
        for (int i = 0; i < 4; i++)
            af[i] = load_a_frag(&As[(wm0 + i * 16 + ln) * 40], kbA);
#pragma unroll
        for (int j = 0; j < 4; j++)
            bfrag[j] = load_b_frag_contig(&Bs[(wn0 + j * 16 + ln) * 40 + kbB]);
#pragma unroll
        for (int i = 0; i < 4; i++)
#pragma unroll
            for (int j = 0; j < 4; j++)
                acc[i][j] = wmma_bf16(af[i], bfrag[j], acc[i][j]);
        __syncthreads();
    }

    const int mb = (lane >> 4) * 8;
#pragma unroll
    for (int i = 0; i < 4; i++) {
#pragma unroll
        for (int j = 0; j < 4; j++) {
            const int col = colg0 + wn0 + j * 16 + ln;
            const float bv = bias[col];
#pragma unroll
            for (int jj = 0; jj < 8; jj++) {
                const int rowm = rowg0 + wm0 + i * 16 + mb + jj;
                const float val = acc[i][j][jj] + bv;
                if constexpr (MODE == 0) {
                    C[(size_t)rowm * N + col] = (OutT)val;
                } else {
                    C[((size_t)((rowm >> 11) * 1024 + col)) * SS + (rowm & (SS - 1))] = (OutT)val;
                }
            }
        }
    }
}

// Flash attention: grid (S/64, H, B), block 128 (4 waves, 16 q rows each).
// Vt is head-transposed V: Vt[(b*1024 + h*64 + hd)*S + s].
// K/V tiles staged by the Tensor Data Mover (double-buffered) when available.
__global__ void __launch_bounds__(128)
attn_kernel(const __bf16* __restrict__ Q, const __bf16* __restrict__ Km,
            const __bf16* __restrict__ Vt, const float* __restrict__ bias,
            const unsigned char* __restrict__ mask, __bf16* __restrict__ ctx) {
    __shared__ __bf16 Kt[2][64 * 80];    // [key][hd]
    __shared__ __bf16 Vts[2][64 * 80];   // [hd][key]
    __shared__ __bf16 Pt[4 * 16 * 80];

    const int t    = threadIdx.x;
    const int lane = t & 31;
    const int wave = t >> 5;
    const int b    = blockIdx.z;
    const int h    = blockIdx.y;
    const int q0   = blockIdx.x * 64 + wave * 16;
    const int ln   = lane & 15;
    const int mb   = (lane >> 4) * 8;
    const int kb8  = (lane >> 4) * 8;
    const int kb16 = (lane >> 4) * 16;
    const float scale = 0.125f;
    const int NIT = SS / 64;

    const __bf16* qrow = Q + ((size_t)b * SS + q0 + ln) * DD + h * HDD;
    const v16bf qa0 = load_a_frag(qrow, kb8);
    const v16bf qa1 = load_a_frag(qrow + 32, kb8);

    v8f o[4];
    float mrun[8], lrun[8];
#pragma unroll
    for (int nf = 0; nf < 4; nf++) o[nf] = v8f_zero();
#pragma unroll
    for (int jj = 0; jj < 8; jj++) { mrun[jj] = -1e30f; lrun[jj] = 0.f; }

#if USE_TDM
    auto issue_pair = [&](int it2, int buf2) {
        const int kt2 = it2 * 64;
        tdm_load_tile_2d((unsigned)(uintptr_t)&Kt[buf2][0],
                         Km + ((size_t)b * SS + kt2) * DD + h * HDD,
                         64u, 64u, (unsigned long long)DD);
        tdm_load_tile_2d((unsigned)(uintptr_t)&Vts[buf2][0],
                         Vt + ((size_t)(b * 1024 + h * HDD)) * SS + kt2,
                         64u, 64u, (unsigned long long)SS);
    };
#endif

    for (int it = 0; it < NIT; ++it) {
        const int kt  = it * 64;
        const int buf = it & 1;

#if USE_TDM
        if (wave == 0) {
            if (it == 0) issue_pair(0, 0);
            if (it + 1 < NIT) {
                issue_pair(it + 1, (it + 1) & 1);     // prefetch next tile pair
                __builtin_amdgcn_s_wait_tensorcnt(2); // current pair retired, next in flight
            } else {
                __builtin_amdgcn_s_wait_tensorcnt(0);
            }
        }
        __syncthreads();
#else
        {
            const int r = t >> 1, half = t & 1;
            const __bf16* gk = Km + ((size_t)b * SS + kt + r) * DD + h * HDD + half * 32;
            const __bf16* gv = Vt + ((size_t)(b * 1024 + h * HDD + r)) * SS + kt + half * 32;
#pragma unroll
            for (int i = 0; i < 4; i++) {
                *(uint4*)&Kt[buf][r * 80 + half * 32 + i * 8]  = *(const uint4*)(gk + i * 8);
                *(uint4*)&Vts[buf][r * 80 + half * 32 + i * 8] = *(const uint4*)(gv + i * 8);
            }
        }
        __syncthreads();
#endif

        // S = Q K^T
        v8f sacc[4];
#pragma unroll
        for (int nf = 0; nf < 4; nf++) {
            const __bf16* krow = &Kt[buf][(nf * 16 + ln) * 80];
            v8f z = v8f_zero();
            z = wmma_bf16(qa0, load_b_frag_contig(krow + kb16), z);
            z = wmma_bf16(qa1, load_b_frag_contig(krow + 32 + kb16), z);
            sacc[nf] = z;
        }

        // scale + bias + key-padding mask
        float sv[4][8];
#pragma unroll
        for (int nf = 0; nf < 4; nf++) {
            const int key = kt + nf * 16 + ln;
            const bool km = mask[b * SS + key] != 0;
            const float* bp = bias + ((size_t)h * SS + (q0 + mb)) * SS + key;
            if (kt + 64 < SS) __builtin_prefetch(bp + 64, 0, 1);
#pragma unroll
            for (int jj = 0; jj < 8; jj++) {
                const float s = sacc[nf][jj] * scale + bp[(size_t)jj * SS];
                sv[nf][jj] = km ? -1e30f : s;
            }
        }

        // online softmax over 16-lane row groups
#pragma unroll
        for (int jj = 0; jj < 8; jj++) {
            float tm = fmaxf(fmaxf(sv[0][jj], sv[1][jj]), fmaxf(sv[2][jj], sv[3][jj]));
            tm = fmaxf(tm, __shfl_xor(tm, 1, 16));
            tm = fmaxf(tm, __shfl_xor(tm, 2, 16));
            tm = fmaxf(tm, __shfl_xor(tm, 4, 16));
            tm = fmaxf(tm, __shfl_xor(tm, 8, 16));
            const float mnew = fmaxf(mrun[jj], tm);
            const float fac  = __expf(mrun[jj] - mnew);
            mrun[jj] = mnew;
            float ps = 0.f;
#pragma unroll
            for (int nf = 0; nf < 4; nf++) {
                const float p = __expf(sv[nf][jj] - mnew);
                sv[nf][jj] = p;
                ps += p;
            }
            ps += __shfl_xor(ps, 1, 16);
            ps += __shfl_xor(ps, 2, 16);
            ps += __shfl_xor(ps, 4, 16);
            ps += __shfl_xor(ps, 8, 16);
            lrun[jj] = lrun[jj] * fac + ps;
#pragma unroll
            for (int nf = 0; nf < 4; nf++) o[nf][jj] *= fac;
        }

        // P: C-layout -> A-layout via per-wave LDS round trip
        __bf16* pw = &Pt[wave * 16 * 80];
#pragma unroll
        for (int nf = 0; nf < 4; nf++)
#pragma unroll
            for (int jj = 0; jj < 8; jj++)
                pw[(mb + jj) * 80 + nf * 16 + ln] = (__bf16)sv[nf][jj];

        const __bf16* prow = pw + ln * 80;
        const v16bf pa0 = load_a_frag(prow, kb8);
        const v16bf pa1 = load_a_frag(prow + 32, kb8);

        // O += P V
#pragma unroll
        for (int nf = 0; nf < 4; nf++) {
            const __bf16* vrow = &Vts[buf][(nf * 16 + ln) * 80];
            o[nf] = wmma_bf16(pa0, load_b_frag_contig(vrow + kb16), o[nf]);
            o[nf] = wmma_bf16(pa1, load_b_frag_contig(vrow + 32 + kb16), o[nf]);
        }
        __syncthreads();
    }

#pragma unroll
    for (int nf = 0; nf < 4; nf++) {
#pragma unroll
        for (int jj = 0; jj < 8; jj++) {
            const float inv = lrun[jj] > 0.f ? 1.f / lrun[jj] : 0.f;
            ctx[((size_t)b * SS + q0 + mb + jj) * DD + h * HDD + nf * 16 + ln] =
                (__bf16)(o[nf][jj] * inv);
        }
    }
}

extern "C" void kernel_launch(void* const* d_in, const int* in_sizes, int n_in,
                              void* d_out, int out_size, void* d_ws, size_t ws_size,
                              hipStream_t stream) {
    (void)in_sizes; (void)n_in; (void)out_size; (void)ws_size;

    const float* x   = (const float*)d_in[0];
    const float* ab  = (const float*)d_in[1];
    const unsigned char* kmask = (const unsigned char*)d_in[2];
    const float* Wq = (const float*)d_in[3];
    const float* bq = (const float*)d_in[4];
    const float* Wk = (const float*)d_in[5];
    const float* bk = (const float*)d_in[6];
    const float* Wv = (const float*)d_in[7];
    const float* bv = (const float*)d_in[8];
    const float* Wo = (const float*)d_in[9];
    const float* bo = (const float*)d_in[10];
    float* out = (float*)d_out;

    const int M = BB * SS;
    char* ws = (char*)d_ws;
    size_t off = 0;
    auto take = [&](size_t bytes) -> char* {
        char* p = ws + off;
        off += (bytes + 255) & ~(size_t)255;
        return p;
    };
    __bf16* xb   = (__bf16*)take((size_t)M * DD * 2);
    __bf16* wqt  = (__bf16*)take((size_t)DD * DD * 2);
    __bf16* wkt  = (__bf16*)take((size_t)DD * DD * 2);
    __bf16* wvt  = (__bf16*)take((size_t)DD * DD * 2);
    __bf16* wot  = (__bf16*)take((size_t)DD * DD * 2);
    __bf16* qb   = (__bf16*)take((size_t)M * DD * 2);
    __bf16* kbuf = (__bf16*)take((size_t)M * DD * 2);
    __bf16* vtb  = (__bf16*)take((size_t)M * DD * 2);
    __bf16* ctxb = (__bf16*)take((size_t)M * DD * 2);

    const int nx = M * DD;
    const int nw = DD * DD;
    f32_to_bf16_kernel<<<(nx + 255) / 256, 256, 0, stream>>>(x, xb, nx);
    transpose_f32_to_bf16_kernel<<<(nw + 255) / 256, 256, 0, stream>>>(Wq, wqt, DD, DD);
    transpose_f32_to_bf16_kernel<<<(nw + 255) / 256, 256, 0, stream>>>(Wk, wkt, DD, DD);
    transpose_f32_to_bf16_kernel<<<(nw + 255) / 256, 256, 0, stream>>>(Wv, wvt, DD, DD);
    transpose_f32_to_bf16_kernel<<<(nw + 255) / 256, 256, 0, stream>>>(Wo, wot, DD, DD);

    dim3 gg(M / 128, DD / 128);
    gemm_bt_bias_kernel<__bf16, 0><<<gg, 128, 0, stream>>>(xb, wqt, bq, qb,   M, DD, DD);
    gemm_bt_bias_kernel<__bf16, 0><<<gg, 128, 0, stream>>>(xb, wkt, bk, kbuf, M, DD, DD);
    gemm_bt_bias_kernel<__bf16, 1><<<gg, 128, 0, stream>>>(xb, wvt, bv, vtb,  M, DD, DD);

    dim3 ga(SS / 64, HH, BB);
    attn_kernel<<<ga, 128, 0, stream>>>(qb, kbuf, vtb, ab, kmask, ctxb);

    gemm_bt_bias_kernel<float, 0><<<gg, 128, 0, stream>>>(ctxb, wot, bo, out, M, DD, DD);
}
